// info_nce_calculation_89481348645584
// MI455X (gfx1250) — compile-verified, matrix operand order checked by reading, not run on hardware
//
#include <hip/hip_runtime.h>
#include <hip/hip_bf16.h>

typedef __attribute__((ext_vector_type(16))) _Float16 v16h;
typedef __attribute__((ext_vector_type(8)))  _Float16 v8h;
typedef __attribute__((ext_vector_type(8)))  float    v8f;

#define DDIM 512           // feature dim (fixed by reference)
#define KCHUNKS (DDIM / 32)
#define NCHUNKS 8          // column chunks for denom partials

// ---------------------------------------------------------------------------
// Kernel A: L2-normalize both projections into f16 reps buffer, compute the
// positive-pair cosine similarity in f32.
// ---------------------------------------------------------------------------
__global__ __launch_bounds__(256)
void normalize_pos_kernel(const float* __restrict__ p1,
                          const float* __restrict__ p2,
                          _Float16* __restrict__ Z,
                          float* __restrict__ pos,
                          int B) {
  const int i = blockIdx.x;
  const int t = threadIdx.x;
  const float* r1 = p1 + (size_t)i * DDIM;
  const float* r2 = p2 + (size_t)i * DDIM;

  const float a0 = r1[t], a1 = r1[t + 256];
  const float b0 = r2[t], b1 = r2[t + 256];

  __shared__ float S1[256], S2[256], SD[256];
  S1[t] = a0 * a0 + a1 * a1;
  S2[t] = b0 * b0 + b1 * b1;
  SD[t] = a0 * b0 + a1 * b1;
  __syncthreads();
  for (int off = 128; off > 0; off >>= 1) {
    if (t < off) { S1[t] += S1[t + off]; S2[t] += S2[t + off]; SD[t] += SD[t + off]; }
    __syncthreads();
  }
  const float n1 = fmaxf(sqrtf(S1[0]), 1e-12f);
  const float n2 = fmaxf(sqrtf(S2[0]), 1e-12f);
  const float inv1 = 1.0f / n1, inv2 = 1.0f / n2;

  _Float16* z1 = Z + (size_t)i * DDIM;
  _Float16* z2 = Z + ((size_t)B + i) * DDIM;
  z1[t]       = (_Float16)(a0 * inv1);
  z1[t + 256] = (_Float16)(a1 * inv1);
  z2[t]       = (_Float16)(b0 * inv2);
  z2[t + 256] = (_Float16)(b1 * inv2);

  if (t == 0) {
    const float p = SD[0] * inv1 * inv2;
    pos[i] = p;
    pos[B + i] = p;
  }
}

// 16x32 f16 operand chunk, CDNA5 16-bit A-matrix layout: lane (l&15) owns row,
// low half-wave K {0..7,16..23}, high half-wave K {8..15,24..31}. Same pattern
// for B since C = Z * Z^T makes both operands rows of Z.
__device__ __forceinline__ v16h load_chunk(const _Float16* __restrict__ row,
                                           int kbase, int hi) {
  const _Float16* p = row + kbase + hi * 8;
  v8h s0 = *(const v8h*)(p);
  v8h s1 = *(const v8h*)(p + 16);
  return __builtin_shufflevector(s0, s1, 0, 1, 2, 3, 4, 5, 6, 7,
                                 8, 9, 10, 11, 12, 13, 14, 15);
}

__device__ __forceinline__ void load_btile(const _Float16* __restrict__ Z,
                                           int jt, int rlo, int hi,
                                           v16h (&Bv)[KCHUNKS]) {
  const _Float16* brow = Z + (size_t)(jt * 16 + rlo) * DDIM;
#pragma unroll
  for (int kc = 0; kc < KCHUNKS; ++kc)
    Bv[kc] = load_chunk(brow, kc * 32, hi);
}

// 16 WMMAs over K=512 with two independent accumulators (matrix-pipe ILP),
// then exp(2*sim) folded into the per-lane row sums.
__device__ __forceinline__ void compute_tile(const v16h (&A)[KCHUNKS],
                                             const v16h (&Bv)[KCHUNKS],
                                             float (&acc)[8]) {
  v8f c0 = {}, c1 = {};
#pragma unroll
  for (int kc = 0; kc < KCHUNKS; kc += 2) {
    c0 = __builtin_amdgcn_wmma_f32_16x16x32_f16(
        false, A[kc],     false, Bv[kc],     (short)0, c0, false, false);
    c1 = __builtin_amdgcn_wmma_f32_16x16x32_f16(
        false, A[kc + 1], false, Bv[kc + 1], (short)0, c1, false, false);
  }
#pragma unroll
  for (int r = 0; r < 8; ++r)
    acc[r] += __expf(2.0f * (c0[r] + c1[r]));   // exp(sim / tau), tau = 0.5
}

// ---------------------------------------------------------------------------
// Kernel B: fused sim-tile GEMM + exp row-sum, double-buffered B tiles.
// grid = (rowTiles/4, NCHUNKS), block = 128 (4 waves, one 16-row tile each).
// ---------------------------------------------------------------------------
__global__ __launch_bounds__(128)
void simexp_rowsum_kernel(const _Float16* __restrict__ Z,
                          float* __restrict__ denomP,
                          int N, int tilesPerChunk) {
  const int wave = threadIdx.x >> 5;
  const int lane = threadIdx.x & 31;
  const int rlo  = lane & 15;
  const int hi   = lane >> 4;

  const int rowTile = blockIdx.x * 4 + wave;
  const int m0 = rowTile * 16;
  const int cchunk = blockIdx.y;

  // A strip: 16 rows x 512 K, persistent in VGPRs (128 VGPRs)
  const _Float16* arow = Z + (size_t)(m0 + rlo) * DDIM;
  v16h A[KCHUNKS];
#pragma unroll
  for (int kc = 0; kc < KCHUNKS; ++kc)
    A[kc] = load_chunk(arow, kc * 32, hi);

  float acc[8];
#pragma unroll
  for (int r = 0; r < 8; ++r) acc[r] = 0.0f;

  const int jt0 = cchunk * tilesPerChunk;
  const int tiles = tilesPerChunk;   // 64: even by construction

  v16h B0[KCHUNKS], B1[KCHUNKS];
  load_btile(Z, jt0, rlo, hi, B0);

  for (int j = 0; j < tiles; j += 2) {
    if (j + 4 < tiles)   // speculative prefetch, 4 tiles ahead
      __builtin_prefetch(Z + (size_t)((jt0 + j + 4) * 16 + rlo) * DDIM, 0, 3);

    load_btile(Z, jt0 + j + 1, rlo, hi, B1);   // in flight during compute(B0)
    compute_tile(A, B0, acc);
    if (j + 2 < tiles)
      load_btile(Z, jt0 + j + 2, rlo, hi, B0); // in flight during compute(B1)
    compute_tile(A, B1, acc);
  }

  // Reduce across the 16 lanes of each half-wave (xor<=8 stays in-group).
#pragma unroll
  for (int r = 0; r < 8; ++r) {
    float v = acc[r];
    v += __shfl_xor(v, 1);
    v += __shfl_xor(v, 2);
    v += __shfl_xor(v, 4);
    v += __shfl_xor(v, 8);
    acc[r] = v;
  }
  if (rlo == 0) {
    // hi=0 lane0 -> rows m0+0..7 ; hi=1 lane16 -> rows m0+8..15
    float* out = denomP + (size_t)cchunk * N + m0 + hi * 8;
#pragma unroll
    for (int r = 0; r < 8; ++r) out[r] = acc[r];
  }
}

// ---------------------------------------------------------------------------
// Kernel C: loss = mean_i( log(sum_j!=i exp(2 sim_ij)) - 2 * pos_i )
// Diagonal removed analytically: denom_i - e^2.
// ---------------------------------------------------------------------------
__global__ __launch_bounds__(256)
void finalize_kernel(const float* __restrict__ denomP,
                     const float* __restrict__ pos,
                     float* __restrict__ out, int N) {
  const int t = threadIdx.x;
  const float diag = 7.38905609893065f;  // exp(2.0)
  float s = 0.0f;
  for (int i = t; i < N; i += 256) {
    float d = 0.0f;
#pragma unroll
    for (int c = 0; c < NCHUNKS; ++c) d += denomP[(size_t)c * N + i];
    d -= diag;
    s += logf(d) - 2.0f * pos[i];
  }
  __shared__ float S[256];
  S[t] = s;
  __syncthreads();
  for (int off = 128; off > 0; off >>= 1) {
    if (t < off) S[t] += S[t + off];
    __syncthreads();
  }
  if (t == 0) out[0] = S[0] / (float)N;
}

extern "C" void kernel_launch(void* const* d_in, const int* in_sizes, int n_in,
                              void* d_out, int out_size, void* d_ws, size_t ws_size,
                              hipStream_t stream) {
  const float* proj1 = (const float*)d_in[0];
  const float* proj2 = (const float*)d_in[1];
  const int B = in_sizes[0] / DDIM;   // 4096
  const int N = 2 * B;                // 8192

  char* ws = (char*)d_ws;
  _Float16* Z = (_Float16*)ws;
  const size_t zbytes = (size_t)N * DDIM * sizeof(_Float16);   // 8 MB
  float* pos    = (float*)(ws + zbytes);
  float* denomP = (float*)(ws + zbytes + (size_t)N * sizeof(float));

  const int rowTiles = N / 16;                       // 512
  const int tilesPerChunk = rowTiles / NCHUNKS;      // 64

  normalize_pos_kernel<<<B, 256, 0, stream>>>(proj1, proj2, Z, pos, B);

  dim3 grid(rowTiles / 4, NCHUNKS);                  // 128 x 8 blocks, 4 waves each
  simexp_rowsum_kernel<<<grid, 128, 0, stream>>>(Z, denomP, N, tilesPerChunk);

  finalize_kernel<<<1, 256, 0, stream>>>(denomP, pos, (float*)d_out, N);
}
